// GNN_84061099917639
// MI455X (gfx1250) — matrix operand order repaired and checked by reading, not compiled, not run on hardware
//
#include <hip/hip_runtime.h>
#include <hip/hip_bf16.h>
#include <cstdint>

typedef __attribute__((ext_vector_type(16))) _Float16 v16h;
typedef __attribute__((ext_vector_type(8)))  float    v8f;

#define IN_D  64
#define HID   128
#define OUT_D 64

// ---------------- degree / normalization ----------------

__global__ void k_deg_init(float* __restrict__ deg, int n) {
  int i = blockIdx.x * blockDim.x + threadIdx.x;
  if (i < n) deg[i] = 1.0f;  // self-loop contribution
}

__global__ void k_degree(const int* __restrict__ dst, float* __restrict__ deg, int e) {
  int i = blockIdx.x * blockDim.x + threadIdx.x;
  if (i < e) atomicAdd(&deg[dst[i]], 1.0f);
}

__global__ void k_dinv(const float* __restrict__ deg, float* __restrict__ dinv, int n) {
  int i = blockIdx.x * blockDim.x + threadIdx.x;
  if (i < n) dinv[i] = rsqrtf(deg[i]);  // deg >= 1 always (self-loops)
}

// ---------------- WMMA GEMM with fused dinv row-scale ----------------
// One wave per 16x16 output tile of HS = (X @ W) * dinv[row].
// W is staged once per block into LDS, transposed + converted to fp16, so each
// lane's 16 B-operand elements (K = half*16+e, fixed column) are one contiguous
// 32-byte LDS read. A is loaded as float4 (K contiguous per 8 elems) + packed cvt.
// Result written to both HS (gather source) and AGG (self-loop init).

template<int K, int NOUT>
__global__ void k_gemm_scale(const float* __restrict__ X, const float* __restrict__ W,
                             const float* __restrict__ dinv,
                             float* __restrict__ HS, float* __restrict__ AGG, int n) {
  __shared__ _Float16 ldsWT[K * NOUT];  // transposed: [col][k], 16KB max

  // cooperative load + transpose + fp32->fp16 convert of W (once per block)
  for (int idx = threadIdx.x; idx < K * NOUT; idx += blockDim.x) {
    int k  = idx / NOUT;
    int nc = idx - k * NOUT;
    ldsWT[nc * K + k] = (_Float16)W[idx];
  }
  __syncthreads();

  const int gt   = blockIdx.x * blockDim.x + threadIdx.x;
  const int wave = gt >> 5;
  const int lane = threadIdx.x & 31;
  const int tilesN = NOUT / 16;
  const int tilesM = (n + 15) / 16;
  if (wave >= tilesM * tilesN) return;          // wave-uniform exit (after barrier)
  const int tm = wave / tilesN;
  const int tn = wave - tm * tilesN;

  const int half = lane >> 4;                   // lane group 0..15 / 16..31
  const int l15  = lane & 15;
  const int m    = tm * 16 + l15;               // A-matrix row for this lane
  const int nc   = tn * 16 + l15;               // B-matrix col for this lane

  v8f c = {};

  if (tm * 16 + 15 < n) {
    // -------- fast path: full tile, branchless --------
    const float4* __restrict__ Xrow = (const float4*)(X + (size_t)m * K);
    #pragma unroll
    for (int k0 = 0; k0 < K; k0 += 32) {
      // A 16x32 fp16: elem e -> K = k0+half*16+e
      float4 p0 = Xrow[(k0 + half * 16) >> 2];
      float4 p1 = Xrow[((k0 + half * 16) >> 2) + 1];
      float4 p2 = Xrow[((k0 + half * 16) >> 2) + 2];
      float4 p3 = Xrow[((k0 + half * 16) >> 2) + 3];
      v16h a;
      a[0] = (_Float16)p0.x;  a[1] = (_Float16)p0.y;
      a[2] = (_Float16)p0.z;  a[3] = (_Float16)p0.w;
      a[4] = (_Float16)p1.x;  a[5] = (_Float16)p1.y;
      a[6] = (_Float16)p1.z;  a[7] = (_Float16)p1.w;
      a[8]  = (_Float16)p2.x; a[9]  = (_Float16)p2.y;
      a[10] = (_Float16)p2.z; a[11] = (_Float16)p2.w;
      a[12] = (_Float16)p3.x; a[13] = (_Float16)p3.y;
      a[14] = (_Float16)p3.z; a[15] = (_Float16)p3.w;
      // B 32x16 fp16: elem e -> K = k0+half*16+e, column nc : contiguous 32B in LDS
      v16h b = *(const v16h*)(ldsWT + nc * K + k0 + half * 16);
      c = __builtin_amdgcn_wmma_f32_16x16x32_f16(
          false, a, false, b, (short)0, c, false, false);
    }
    // epilogue: VGPR r holds row = tm*16 + half*8 + r, column nc — branchless
    #pragma unroll
    for (int r = 0; r < 8; ++r) {
      int row = tm * 16 + half * 8 + r;
      float v = c[r] * dinv[row];
      size_t off = (size_t)row * NOUT + nc;
      HS[off]  = v;   // gather source
      AGG[off] = v;   // self-loop init of aggregation
    }
  } else {
    // -------- tail path: partially valid tile (not hit when 16 | n) --------
    const bool mvalid = (m < n);
    const int  mr = mvalid ? m : (n - 1);       // clamp to avoid OOB reads
    #pragma unroll
    for (int k0 = 0; k0 < K; k0 += 32) {
      v16h a, b;
      #pragma unroll
      for (int e = 0; e < 8; ++e) {
        float a0 = X[(size_t)mr * K + k0 + half * 16 + e];
        float a1 = X[(size_t)mr * K + k0 + half * 16 + 8 + e];
        a[e]     = (_Float16)(mvalid ? a0 : 0.0f);
        a[e + 8] = (_Float16)(mvalid ? a1 : 0.0f);
      }
      b = *(const v16h*)(ldsWT + nc * K + k0 + half * 16);
      c = __builtin_amdgcn_wmma_f32_16x16x32_f16(
          false, a, false, b, (short)0, c, false, false);
    }
    #pragma unroll
    for (int r = 0; r < 8; ++r) {
      int row = tm * 16 + half * 8 + r;
      if (row < n) {
        float v = c[r] * dinv[row];
        size_t off = (size_t)row * NOUT + nc;
        HS[off]  = v;
        AGG[off] = v;
      }
    }
  }
}

// ---------------- edge scatter-add (wave per edge) ----------------

template<int F>
__global__ void k_scatter(const int* __restrict__ src, const int* __restrict__ dst,
                          const float* __restrict__ hs, float* __restrict__ agg, int e) {
  const int gt   = blockIdx.x * blockDim.x + threadIdx.x;
  const int edge = gt >> 5;
  const int lane = threadIdx.x & 31;
  if (edge >= e) return;
  const int s = src[edge];
  const int d = dst[edge];
  const float* __restrict__ hrow = hs + (size_t)s * F;
  float* arow = agg + (size_t)d * F;
  #pragma unroll
  for (int f = lane; f < F; f += 32) {
    atomicAdd(&arow[f], hrow[f]);   // global_atomic_add_f32
  }
}

// ---------------- fused dinv * agg + bias, ReLU (float4) ----------------

template<int F>
__global__ void k_post_relu(const float4* __restrict__ agg, const float* __restrict__ dinv,
                            const float* __restrict__ b, float4* __restrict__ out, int n) {
  int i = blockIdx.x * blockDim.x + threadIdx.x;      // index in float4 units
  int total4 = n * (F / 4);
  if (i >= total4) return;
  int node = i / (F / 4);
  int col4 = i - node * (F / 4);
  float d = dinv[node];
  float4 v = agg[i];
  const float* bb = b + col4 * 4;
  v.x = v.x * d + bb[0];  v.y = v.y * d + bb[1];
  v.z = v.z * d + bb[2];  v.w = v.w * d + bb[3];
  v.x = v.x > 0.0f ? v.x : 0.0f;
  v.y = v.y > 0.0f ? v.y : 0.0f;
  v.z = v.z > 0.0f ? v.z : 0.0f;
  v.w = v.w > 0.0f ? v.w : 0.0f;
  out[i] = v;
}

// ---------------- final layer: scale+bias+ReLU fused with the two 64->1 heads ----------------
// One wave per node; each lane handles 2 features; wave32 shuffle reduction.

__global__ void k_post2_heads(const float* __restrict__ agg, const float* __restrict__ dinv,
                              const float* __restrict__ b2,
                              const float* __restrict__ Wd, const float* __restrict__ bd,
                              const float* __restrict__ Wp, const float* __restrict__ bp,
                              float* __restrict__ out, int n) {
  const int gt   = blockIdx.x * blockDim.x + threadIdx.x;
  const int node = gt >> 5;
  const int lane = threadIdx.x & 31;
  if (node >= n) return;
  const float di = dinv[node];
  float f0 = agg[(size_t)node * OUT_D + lane]      * di + b2[lane];
  float f1 = agg[(size_t)node * OUT_D + 32 + lane] * di + b2[32 + lane];
  f0 = f0 > 0.0f ? f0 : 0.0f;
  f1 = f1 > 0.0f ? f1 : 0.0f;
  float ds = f0 * Wd[lane] + f1 * Wd[32 + lane];
  float ps = f0 * Wp[lane] + f1 * Wp[32 + lane];
  #pragma unroll
  for (int off = 16; off > 0; off >>= 1) {
    ds += __shfl_xor(ds, off, 32);
    ps += __shfl_xor(ps, off, 32);
  }
  if (lane == 0) {
    out[node]     = ds + bd[0];   // duration_out
    out[n + node] = ps + bp[0];   // phase_out
  }
}

// ---------------- launch ----------------

extern "C" void kernel_launch(void* const* d_in, const int* in_sizes, int n_in,
                              void* d_out, int out_size, void* d_ws, size_t ws_size,
                              hipStream_t stream) {
  const float* x  = (const float*)d_in[0];
  const int*   ei = (const int*)  d_in[1];
  const float* W1 = (const float*)d_in[2];
  const float* b1 = (const float*)d_in[3];
  const float* W2 = (const float*)d_in[4];
  const float* b2 = (const float*)d_in[5];
  const float* Wd = (const float*)d_in[6];
  const float* bd = (const float*)d_in[7];
  const float* Wp = (const float*)d_in[8];
  const float* bp = (const float*)d_in[9];

  const int n = in_sizes[0] / IN_D;   // 100000
  const int e = in_sizes[1] / 2;      // 1600000
  const int* src = ei;
  const int* dst = ei + e;

  // workspace carving (with reuse):
  //   B0: h1s, then reused for relu(layer1) output (gemm2 input)
  //   B1: agg1, then reused for h2s
  //   B2: agg2
  auto align256 = [](size_t o) { return (o + 255) & ~(size_t)255; };
  char* ws = (char*)d_ws;
  size_t o = 0;
  float* deg  = (float*)(ws + o); o = align256(o + (size_t)n * 4);
  float* dinv = (float*)(ws + o); o = align256(o + (size_t)n * 4);
  float* B0   = (float*)(ws + o); o = align256(o + (size_t)n * HID * 4);
  float* B1   = (float*)(ws + o); o = align256(o + (size_t)n * HID * 4);
  float* B2   = (float*)(ws + o); o = align256(o + (size_t)n * OUT_D * 4);
  (void)ws_size; (void)n_in; (void)out_size;

  const int T = 256;

  // normalization factors
  k_deg_init<<<(n + T - 1) / T, T, 0, stream>>>(deg, n);
  k_degree  <<<(e + T - 1) / T, T, 0, stream>>>(dst, deg, e);
  k_dinv    <<<(n + T - 1) / T, T, 0, stream>>>(deg, dinv, n);

  // ---- layer 1: h1s = (x@W1)*dinv ; agg1 = h1s + scatter ; out1 = relu(agg1*dinv+b1)
  {
    int waves  = ((n + 15) / 16) * (HID / 16);
    int blocks = (waves * 32 + T - 1) / T;
    k_gemm_scale<IN_D, HID><<<blocks, T, 0, stream>>>(x, W1, dinv, B0, B1, n);

    int sblocks = (e * 32 + T - 1) / T;
    k_scatter<HID><<<sblocks, T, 0, stream>>>(src, dst, B0, B1, e);

    int total4 = n * (HID / 4);
    k_post_relu<HID><<<(total4 + T - 1) / T, T, 0, stream>>>(
        (const float4*)B1, dinv, b1, (float4*)B0, n);
  }

  // ---- layer 2: h2s = (out1@W2)*dinv ; agg2 = h2s + scatter ; heads fused
  {
    int waves  = ((n + 15) / 16) * (OUT_D / 16);
    int blocks = (waves * 32 + T - 1) / T;
    k_gemm_scale<HID, OUT_D><<<blocks, T, 0, stream>>>(B0, W2, dinv, B1, B2, n);

    int sblocks = (e * 32 + T - 1) / T;
    k_scatter<OUT_D><<<sblocks, T, 0, stream>>>(src, dst, B1, B2, e);

    int hblocks = (n * 32 + T - 1) / T;
    k_post2_heads<<<hblocks, T, 0, stream>>>(B2, dinv, b2, Wd, bd, Wp, bp,
                                             (float*)d_out, n);
  }
}
